// LinearAttention_481036337443
// MI455X (gfx1250) — matrix operand order repaired and checked by reading, not compiled, not run on hardware
//
#include <hip/hip_runtime.h>
#include <hip/hip_bf16.h>
#include <math.h>

// Problem constants (from reference)
#define B_   16
#define C_   256      // channels (LN dim, GEMM K for qkv)
#define H_   8
#define D_   64
#define HID  512      // heads*dim_head
#define O3   1536     // 3*HID rows of qkv
#define N_   4096     // tokens (64*64)

typedef __attribute__((ext_vector_type(16))) __bf16 v16bf;
typedef __attribute__((ext_vector_type(8)))  float  v8f;
typedef __attribute__((ext_vector_type(4)))  unsigned int v4u;
typedef __attribute__((ext_vector_type(8)))  int v8i;
typedef __attribute__((ext_vector_type(4)))  int v4i;

union Frag16 { v16bf v; unsigned short u[16]; };

// Native bf16 convert: backend emits v_cvt_(pk_)bf16_f32 on gfx1250.
__device__ __forceinline__ unsigned short f2bf(float f) {
  __bf16 h = (__bf16)f;
  return __builtin_bit_cast(unsigned short, h);
}
__device__ __forceinline__ float bf2f(unsigned short s) {
  return __uint_as_float(((unsigned int)s) << 16);
}

// ---------------------------------------------------------------------------
// K1: qkv[b][o][n] = sum_c w_qkv[o][c] * x[b][c][n]  (bf16 out, f32 acc WMMA)
// block = 256 thr = 8 waves (4 M x 2 N), block tile 64x64, K step 32.
// Staging uses b128 global loads; A converts+packs to bf16 pairs (b64 LDS).
// ---------------------------------------------------------------------------
__global__ void qkv_gemm(const float* __restrict__ x,
                         const float* __restrict__ wqkv,
                         unsigned short* __restrict__ qkv) {
  __shared__ unsigned short As[64][36];   // A tile [m][k]
  __shared__ unsigned short Bs[64][36];   // B tile transposed [n][k]
  const int t = threadIdx.x, lane = t & 31, wave = t >> 5;
  const int wm = wave >> 1, wn = wave & 1;
  const int n0 = blockIdx.x * 64, m0 = blockIdx.y * 64, b = blockIdx.z;
  const float* xb = x + (size_t)b * C_ * N_;
  const int row = lane & 15, kbA = (lane >> 4) * 8, kbB = (lane >> 4) * 16;
  v8f c0 = {}; v8f c1 = {};
  for (int k0 = 0; k0 < C_; k0 += 32) {
#pragma unroll
    for (int i = 0; i < 2; ++i) {                 // A: 64x32 = 512 float4
      int idx = t + i * 256;
      int m = idx >> 3, k4 = (idx & 7) * 4;
      float4 v = *(const float4*)&wqkv[(size_t)(m0 + m) * C_ + k0 + k4];
      uint2 p;
      p.x = (unsigned)f2bf(v.x) | ((unsigned)f2bf(v.y) << 16);
      p.y = (unsigned)f2bf(v.z) | ((unsigned)f2bf(v.w) << 16);
      *(uint2*)&As[m][k4] = p;
    }
#pragma unroll
    for (int i = 0; i < 2; ++i) {                 // B: 32x64 = 512 float4, transpose
      int idx = t + i * 256;
      int k = idx >> 4, n4 = (idx & 15) * 4;
      float4 v = *(const float4*)&xb[(size_t)(k0 + k) * N_ + n0 + n4];
      Bs[n4 + 0][k] = f2bf(v.x);
      Bs[n4 + 1][k] = f2bf(v.y);
      Bs[n4 + 2][k] = f2bf(v.z);
      Bs[n4 + 3][k] = f2bf(v.w);
    }
    if (k0 + 32 < C_) {                           // prefetch next K tiles
      __builtin_prefetch(&wqkv[(size_t)(m0 + (t >> 2)) * C_ + k0 + 32], 0, 1);
      __builtin_prefetch(&xb[(size_t)(k0 + 32 + (t >> 3)) * N_ + n0], 0, 1);
    }
    __syncthreads();
    Frag16 a, b0, b1;
#pragma unroll
    for (int e = 0; e < 16; ++e) {
      a.u[e]  = As[wm * 16 + row][kbB + e];
      b0.u[e] = Bs[wn * 32 + row][kbB + e];
      b1.u[e] = Bs[wn * 32 + 16 + row][kbB + e];
    }
    c0 = __builtin_amdgcn_wmma_f32_16x16x32_bf16(false, a.v, false, b0.v, (short)0, c0, false, false);
    c1 = __builtin_amdgcn_wmma_f32_16x16x32_bf16(false, a.v, false, b1.v, (short)0, c1, false, false);
    __syncthreads();
  }
  unsigned short* out = qkv + (size_t)b * O3 * N_;
  const int mh = (lane >> 4) * 8;
#pragma unroll
  for (int r = 0; r < 8; ++r) {
    int o = m0 + wm * 16 + r + mh;
    out[(size_t)o * N_ + n0 + wn * 32 + row]      = f2bf(c0[r]);
    out[(size_t)o * N_ + n0 + wn * 32 + 16 + row] = f2bf(c1[r]);
  }
}

// ---------------------------------------------------------------------------
// K2: in-place softmax of q over d (64 vals), fold scale=1/8. 1 thread/column.
// ---------------------------------------------------------------------------
__global__ void q_softmax(unsigned short* __restrict__ qkv) {
  int g = blockIdx.x * blockDim.x + threadIdx.x;   // 16*8*4096 threads
  int n = g & (N_ - 1);
  int h = (g >> 12) & 7;
  int b = g >> 15;
  unsigned short* base = qkv + (size_t)b * O3 * N_ + (size_t)(h * 64) * N_ + n;
  unsigned short qv[64];
  float mx = -3.0e38f;
#pragma unroll
  for (int d = 0; d < 64; ++d) { qv[d] = base[(size_t)d * N_]; mx = fmaxf(mx, bf2f(qv[d])); }
  float s = 0.f;
#pragma unroll
  for (int d = 0; d < 64; ++d) s += __expf(bf2f(qv[d]) - mx);
  float inv = 0.125f / s;                          // softmax * dim_head^-0.5
#pragma unroll
  for (int d = 0; d < 64; ++d)
    base[(size_t)d * N_] = f2bf(__expf(bf2f(qv[d]) - mx) * inv);
}

// ---------------------------------------------------------------------------
// K3: per-row stats of k over n=4096: kmax, 1/(sumexp * n). 1 block/row.
// ---------------------------------------------------------------------------
__global__ void k_stats(const unsigned short* __restrict__ qkv,
                        float* __restrict__ kmax, float* __restrict__ kinv) {
  __shared__ float red[256];
  int rowg = blockIdx.x;                 // (b*8+h)*64 + d  (8192 rows)
  int b = rowg >> 9, r = rowg & 511;
  const unsigned short* base = qkv + (size_t)b * O3 * N_ + (size_t)(512 + r) * N_;
  int t = threadIdx.x;
  float m = -3.0e38f;
  for (int i = t; i < N_; i += 256) m = fmaxf(m, bf2f(base[i]));
  red[t] = m; __syncthreads();
  for (int st = 128; st > 0; st >>= 1) { if (t < st) red[t] = fmaxf(red[t], red[t + st]); __syncthreads(); }
  float rowm = red[0]; __syncthreads();
  float s = 0.f;
  for (int i = t; i < N_; i += 256) s += __expf(bf2f(base[i]) - rowm);
  red[t] = s; __syncthreads();
  for (int st = 128; st > 0; st >>= 1) { if (t < st) red[t] += red[t + st]; __syncthreads(); }
  if (t == 0) { kmax[rowg] = rowm; kinv[rowg] = 1.f / (red[0] * (float)N_); }
}

// ---------------------------------------------------------------------------
// K4: ctx[b,h,d,e] = sum_n softmax(k)[d,n] * (v[e,n]/n). One block per (b,h).
// 512 thr = 16 waves, 4x4 grid of 16x16 tiles.
// V tile staged by the Tensor Data Mover (TDM): 2D D# descriptor with
// LDS padding (64B data + 8B pad per row -> matches Vs[64][36]); synced via
// s_wait_tensorcnt. K tile staged manually (exp transform applied).
// ---------------------------------------------------------------------------
__global__ void context_gemm(const unsigned short* __restrict__ qkv,
                             const float* __restrict__ kmax,
                             const float* __restrict__ kinv,
                             float* __restrict__ ctx) {
  __shared__ unsigned short Ks[64][36];
  __shared__ unsigned short Vs[64][36];
  int h = blockIdx.x, b = blockIdx.y;
  int t = threadIdx.x, lane = t & 31, wave = t >> 5;
  int wi = wave >> 2, wj = wave & 3;
  const unsigned short* kb_ = qkv + (size_t)b * O3 * N_ + (size_t)(512 + h * 64) * N_;
  const unsigned short* vb_ = qkv + (size_t)b * O3 * N_ + (size_t)(1024 + h * 64) * N_;
  int rb = (b * 8 + h) * 64;
  int row = lane & 15, kbA = (lane >> 4) * 8, kbB = (lane >> 4) * 16;
  const unsigned vs_lds = (unsigned)(uintptr_t)&Vs[0][0];
  v8f c = {};
  for (int n0 = 0; n0 < N_; n0 += 32) {
    if (wave == 0) {
      // D# group0: count=1 | lds_addr | global_addr(57b) | type=2
      unsigned long long ga = (unsigned long long)(uintptr_t)(vb_ + n0);
      v4u g0;
      g0[0] = 1u;
      g0[1] = vs_lds;
      g0[2] = (unsigned)ga;
      g0[3] = ((unsigned)(ga >> 32) & 0x01FFFFFFu) | 0x80000000u;
      // D# group1: data_size=2B, pad_enable, pad_interval=16DW, pad_amount=2DW,
      // tensor_dim0=4096, tensor_dim1=64, tile_dim0=32, tile_dim1=64, stride0=4096
      v8i g1;
      g1[0] = 0x02D10000;          // mask=0 | data_size=1<<16 | pad bits
      g1[1] = (int)(4096u << 16);  // tensor_dim0[15:0]
      g1[2] = (int)(64u << 16);    // tensor_dim0 hi=0 | tensor_dim1[15:0]
      g1[3] = (int)(32u << 16);    // tensor_dim1 hi=0 | tile_dim0
      g1[4] = 64;                  // tile_dim1 | tile_dim2=0
      g1[5] = 4096;                // tensor_dim0_stride[31:0]
      g1[6] = 0;                   // stride0 hi | stride1 lo
      g1[7] = 0;                   // stride1 hi
      v4i gz = {0, 0, 0, 0};
      v8i gz8 = {0, 0, 0, 0, 0, 0, 0, 0};
      __builtin_amdgcn_tensor_load_to_lds(g0, g1, gz, gz, gz8, 0);
    }
#pragma unroll
    for (int i = 0; i < 4; ++i) {            // K tile with exp applied
      int idx = t + i * 512, d = idx >> 5, nn = idx & 31;
      float kv = bf2f(kb_[(size_t)d * N_ + n0 + nn]);
      Ks[d][nn] = f2bf(__expf(kv - kmax[rb + d]) * kinv[rb + d]);
    }
    if (wave == 0) __builtin_amdgcn_s_wait_tensorcnt(0);
    __syncthreads();
    Frag16 a, bb;
#pragma unroll
    for (int e = 0; e < 16; ++e) {
      a.u[e]  = Ks[wi * 16 + row][kbB + e];
      bb.u[e] = Vs[wj * 16 + row][kbB + e];
    }
    c = __builtin_amdgcn_wmma_f32_16x16x32_bf16(false, a.v, false, bb.v, (short)0, c, false, false);
    __syncthreads();
  }
  float* cb = ctx + (size_t)(b * 8 + h) * 64 * 64;
  int mh = (lane >> 4) * 8;
#pragma unroll
  for (int r = 0; r < 8; ++r)
    cb[(wi * 16 + r + mh) * 64 + wj * 16 + row] = c[r];
}

// ---------------------------------------------------------------------------
// K5: w_eff[b][o][h*64+d] = sum_e w_out[o][h*64+e] * ctx[b,h,d,e]  (bf16 out)
// grid (mtile=16, h, b); 128 thr = 4 waves across d-tiles; K=64 (2 steps).
// ---------------------------------------------------------------------------
__global__ void weff_gemm(const float* __restrict__ wout,
                          const float* __restrict__ ctx,
                          unsigned short* __restrict__ weff) {
  int mt = blockIdx.x, h = blockIdx.y, b = blockIdx.z;
  int t = threadIdx.x, lane = t & 31, wj = t >> 5;
  int row = lane & 15, kbA = (lane >> 4) * 8, kbB = (lane >> 4) * 16;
  const float* cb = ctx + (size_t)(b * 8 + h) * 64 * 64;
  v8f c = {};
  for (int k0 = 0; k0 < 64; k0 += 32) {
    Frag16 a, bb;
#pragma unroll
    for (int e = 0; e < 16; ++e) {
      int ke = k0 + kbB + e;
      a.u[e]  = f2bf(wout[(size_t)(mt * 16 + row) * HID + h * 64 + ke]);
      bb.u[e] = f2bf(cb[(wj * 16 + row) * 64 + k0 + kbB + e]);
    }
    c = __builtin_amdgcn_wmma_f32_16x16x32_bf16(false, a.v, false, bb.v, (short)0, c, false, false);
  }
  unsigned short* wb = weff + (size_t)b * C_ * HID;
  int mh = (lane >> 4) * 8;
#pragma unroll
  for (int r = 0; r < 8; ++r)
    wb[(size_t)(mt * 16 + r + mh) * HID + h * 64 + wj * 16 + row] = f2bf(c[r]);
}

// ---------------------------------------------------------------------------
// K6: y[b][o][n] = sum_hd w_eff[b][o][hd] * q_soft[b][hd][n] + b_out[o]
// A,B already bf16 in workspace (q_soft = rows 0..511 of qkv buffer).
// Staging uses b128/b64 global + LDS vector ops (A row pitch 40 => 16B align).
// ---------------------------------------------------------------------------
__global__ void out_gemm(const unsigned short* __restrict__ weff,
                         const unsigned short* __restrict__ qkv,
                         const float* __restrict__ bout,
                         float* __restrict__ y) {
  __shared__ unsigned short As[64][40];
  __shared__ unsigned short Bs[64][36];
  int t = threadIdx.x, lane = t & 31, wave = t >> 5;
  int wm = wave >> 1, wn = wave & 1;
  int n0 = blockIdx.x * 64, m0 = blockIdx.y * 64, b = blockIdx.z;
  const unsigned short* A  = weff + (size_t)b * C_ * HID;
  const unsigned short* Bq = qkv + (size_t)b * O3 * N_;
  int row = lane & 15, kbA = (lane >> 4) * 8, kbB = (lane >> 4) * 16;
  v8f c0 = {}, c1 = {};
  for (int k0 = 0; k0 < HID; k0 += 32) {
    {                                            // A: 64x32 bf16 = 256 x b128
      int m = t >> 2, k8 = (t & 3) * 8;
      *(uint4*)&As[m][k8] = *(const uint4*)&A[(size_t)(m0 + m) * HID + k0 + k8];
    }
#pragma unroll
    for (int i = 0; i < 2; ++i) {                // B: 32x64 bf16 = 512 x b64, transpose
      int idx = t + i * 256;
      int k = idx >> 4, n4 = (idx & 15) * 4;
      uint2 p = *(const uint2*)&Bq[(size_t)(k0 + k) * N_ + n0 + n4];
      Bs[n4 + 0][k] = (unsigned short)(p.x);
      Bs[n4 + 1][k] = (unsigned short)(p.x >> 16);
      Bs[n4 + 2][k] = (unsigned short)(p.y);
      Bs[n4 + 3][k] = (unsigned short)(p.y >> 16);
    }
    if (k0 + 32 < HID)
      __builtin_prefetch(&Bq[(size_t)(k0 + 32 + (t >> 3)) * N_ + n0], 0, 1);
    __syncthreads();
    Frag16 a, b0, b1;
#pragma unroll
    for (int e = 0; e < 16; ++e) {
      a.u[e]  = As[wm * 16 + row][kbB + e];
      b0.u[e] = Bs[wn * 32 + row][kbB + e];
      b1.u[e] = Bs[wn * 32 + 16 + row][kbB + e];
    }
    c0 = __builtin_amdgcn_wmma_f32_16x16x32_bf16(false, a.v, false, b0.v, (short)0, c0, false, false);
    c1 = __builtin_amdgcn_wmma_f32_16x16x32_bf16(false, a.v, false, b1.v, (short)0, c1, false, false);
    __syncthreads();
  }
  int mh = (lane >> 4) * 8;
#pragma unroll
  for (int r = 0; r < 8; ++r) {
    int o = m0 + wm * 16 + r + mh;
    float bo = bout[o];
    size_t rowoff = (size_t)b * C_ * N_ + (size_t)o * N_ + n0;
    y[rowoff + wn * 32 + row]      = c0[r] + bo;
    y[rowoff + wn * 32 + 16 + row] = c1[r] + bo;
  }
}

// ---------------------------------------------------------------------------
// K7: in-place LayerNorm over channels (biased var), *g. 1 thread/column.
// ---------------------------------------------------------------------------
__global__ void layer_norm(float* __restrict__ y, const float* __restrict__ g) {
  int col = blockIdx.x * blockDim.x + threadIdx.x;   // 16*4096 columns
  int b = col >> 12, n = col & (N_ - 1);
  float* base = y + (size_t)b * C_ * N_ + n;
  float s = 0.f, ss = 0.f;
  for (int cc = 0; cc < C_; ++cc) { float v = base[(size_t)cc * N_]; s += v; ss += v * v; }
  float mean = s * (1.f / C_);
  float var  = ss * (1.f / C_) - mean * mean;
  float inv  = rsqrtf(var + 1e-5f);
  for (int cc = 0; cc < C_; ++cc) {
    float v = base[(size_t)cc * N_];
    base[(size_t)cc * N_] = (v - mean) * inv * g[cc];
  }
}

// ---------------------------------------------------------------------------
extern "C" void kernel_launch(void* const* d_in, const int* in_sizes, int n_in,
                              void* d_out, int out_size, void* d_ws, size_t ws_size,
                              hipStream_t stream) {
  const float* x    = (const float*)d_in[0];
  const float* wqkv = (const float*)d_in[1];
  const float* wout = (const float*)d_in[2];
  const float* bout = (const float*)d_in[3];
  const float* g    = (const float*)d_in[4];

  char* ws = (char*)d_ws;
  unsigned short* qkv = (unsigned short*)ws;                 // 16*1536*4096 bf16 = 201.3 MB
  size_t off = (size_t)B_ * O3 * N_ * 2;
  float* ctx = (float*)(ws + off);  off += (size_t)B_ * H_ * D_ * D_ * 4;   // 2 MB
  unsigned short* weff = (unsigned short*)(ws + off); off += (size_t)B_ * C_ * HID * 2; // 4 MB
  float* kmax = (float*)(ws + off); off += 8192 * 4;
  float* kinv = (float*)(ws + off);
  float* y = (float*)d_out;

  qkv_gemm    <<<dim3(64, 24, 16), 256, 0, stream>>>(x, wqkv, qkv);
  q_softmax   <<<2048, 256, 0, stream>>>(qkv);
  k_stats     <<<8192, 256, 0, stream>>>(qkv, kmax, kinv);
  context_gemm<<<dim3(8, 16, 1), 512, 0, stream>>>(qkv, kmax, kinv, ctx);
  weff_gemm   <<<dim3(16, 8, 16), 128, 0, stream>>>(wout, ctx, weff);
  out_gemm    <<<dim3(64, 4, 16), 256, 0, stream>>>(weff, qkv, bout, y);
  layer_norm  <<<256, 256, 0, stream>>>(y, g);
}
